// VideoEncoder_5377299055160
// MI455X (gfx1250) — matrix operand order repaired and checked by reading, not compiled, not run on hardware
//
#include <hip/hip_runtime.h>
#include <hip/hip_bf16.h>

// ---------------- problem constants (match setup_inputs) ----------------
#define NB   2
#define SEQ  2048
#define DM   768
#define NH   8
#define HD   96
#define TD   2304   // 3*DM
#define FF   3072
#define NE   4
#define NL   2
#define NT   (NB*SEQ)   // 4096 tokens

typedef __attribute__((ext_vector_type(16))) __bf16 v16bf;
typedef __attribute__((ext_vector_type(8)))  __bf16 v8bf;
typedef __attribute__((ext_vector_type(8)))  float  v8f;
typedef __attribute__((__vector_size__(4 * sizeof(int)))) int v4i;

// ---- CDNA5 async global->LDS copy (ASYNCcnt path), with safe fallback ----
#if __has_builtin(__builtin_amdgcn_global_load_async_to_lds_b128)
#define ASYNC_CP(g, l)                                                        \
  __builtin_amdgcn_global_load_async_to_lds_b128((v4i*)(g), (v4i*)(l), 0, 0)
#else
#define ASYNC_CP(g, l) (*(v8bf*)(l) = *(const v8bf*)(g))
#endif

#if __has_builtin(__builtin_amdgcn_s_wait_asynccnt)
#define WAIT_ASYNC(n) __builtin_amdgcn_s_wait_asynccnt(n)
#else
#define WAIT_ASYNC(n) asm volatile("s_wait_asynccnt %0" :: "i"(n) : "memory")
#endif

// fp32 -> bf16 round-to-nearest-even (bit math, no intrinsics needed)
static __device__ __forceinline__ __bf16 f2bf(float f){
  unsigned u = __float_as_uint(f);
  u += 0x7fffu + ((u >> 16) & 1u);
  unsigned short s = (unsigned short)(u >> 16);
  __bf16 r;
  __builtin_memcpy(&r, &s, sizeof(r));
  return r;
}

// ---- WMMA fragment loads per CDNA5 ISA 7.12.2 (wave32, 16-bit operands) ----
static __device__ __forceinline__ v16bf frag_a_ld(const __bf16* base, int lda){
  int lane = threadIdx.x & 31;
  int row  = lane & 15;
  int kb   = (lane < 16) ? 0 : 8;
  const __bf16* p = base + row * lda + kb;
  v8bf lo = *(const v8bf*)(p);
  v8bf hi = *(const v8bf*)(p + 16);
  v16bf r;
#pragma unroll
  for (int e = 0; e < 8; ++e){ r[e] = lo[e]; r[e + 8] = hi[e]; }
  return r;
}

static __device__ __forceinline__ v16bf frag_b_ld(const __bf16* base, int ldk){
  int lane = threadIdx.x & 31;
  int col  = lane & 15;
  int kb   = (lane < 16) ? 0 : 8;
  const __bf16* p = base + col * ldk + kb;
  v8bf lo = *(const v8bf*)(p);
  v8bf hi = *(const v8bf*)(p + 16);
  v16bf r;
#pragma unroll
  for (int e = 0; e < 8; ++e){ r[e] = lo[e]; r[e + 8] = hi[e]; }
  return r;
}

static __device__ __forceinline__ v8f wmma_bf16(v16bf a, v16bf b, v8f c){
  return __builtin_amdgcn_wmma_f32_16x16x32_bf16(false, a, false, b, (short)0, c, false, false);
}

// =====================================================================
// GEMM: C[M,N] (+)= rowScale[m] * (A[M,K] @ B[K,N]), optional bf16 dup of C.
// A: bf16 [M,K].  Bt: bf16 pre-transposed [N,K].  C: f32.
// Tile 128x128x32, 8 waves (4 M-strips x 2 N-strips), wave = 32x64 out,
// double-buffered LDS staged with async global->LDS b128 copies.
// M%128==0, N%128==0, K%32==0.
// =====================================================================
__global__ __launch_bounds__(256)
void gemm_kernel(const __bf16* __restrict__ A, const __bf16* __restrict__ Bt,
                 float* __restrict__ C, __bf16* __restrict__ Cbf,
                 const float* __restrict__ rowScale,
                 int M, int N, int K, int scaleStride, int accumulate)
{
  __shared__ __align__(16) __bf16 As[2][128 * 32];   // [m][k]
  __shared__ __align__(16) __bf16 Bs[2][128 * 32];   // [n][k]
  const int tid  = threadIdx.x;
  const int wave = tid >> 5;
  const int lane = tid & 31;
  const int waveM = wave & 3;       // 0..3  -> 32-row strip
  const int waveN = wave >> 2;      // 0..1  -> 64-col strip
  const int rowBase = blockIdx.y * 128;
  const int colBase = blockIdx.x * 128;

  v8f acc[2][4];
#pragma unroll
  for (int i = 0; i < 2; ++i)
#pragma unroll
    for (int g = 0; g < 4; ++g)
#pragma unroll
      for (int e = 0; e < 8; ++e) acc[i][g][e] = 0.0f;

  // stage one 128x32 A tile + 128x32 B tile: 1024 b128 chunks, 4 per thread
  auto stage = [&](int kt, int s){
    const __bf16* Ag = A  + (size_t)rowBase * K + (size_t)kt * 32;
    const __bf16* Bg = Bt + (size_t)colBase * K + (size_t)kt * 32;
#pragma unroll
    for (int j = 0; j < 2; ++j){
      int c = tid + j * 256;          // 0..511
      int row = c >> 2;               // 4 chunks per 32-wide row
      int kc  = (c & 3) * 8;
      ASYNC_CP(Ag + (size_t)row * K + kc, &As[s][row * 32 + kc]);
      ASYNC_CP(Bg + (size_t)row * K + kc, &Bs[s][row * 32 + kc]);
    }
  };

  auto compute = [&](int s){
    v16bf af[2], bfv[4];
#pragma unroll
    for (int i = 0; i < 2; ++i)
      af[i] = frag_a_ld(&As[s][(waveM * 32 + i * 16) * 32], 32);
#pragma unroll
    for (int g = 0; g < 4; ++g)
      bfv[g] = frag_b_ld(&Bs[s][(waveN * 64 + g * 16) * 32], 32);
#pragma unroll
    for (int i = 0; i < 2; ++i)
#pragma unroll
      for (int g = 0; g < 4; ++g)
        acc[i][g] = wmma_bf16(af[i], bfv[g], acc[i][g]);
  };

  const int nk = K >> 5;
  stage(0, 0);
  for (int t = 0; t < nk - 1; ++t){
    stage(t + 1, (t + 1) & 1);
    WAIT_ASYNC(4);                 // allow the 4 just-issued; tile t is done
    __syncthreads();
    compute(t & 1);
    __syncthreads();
  }
  WAIT_ASYNC(0);
  __syncthreads();
  compute((nk - 1) & 1);

  const int halfAdd = (lane < 16) ? 0 : 8;
  const int col0    = lane & 15;
#pragma unroll
  for (int i = 0; i < 2; ++i){
#pragma unroll
    for (int g = 0; g < 4; ++g){
#pragma unroll
      for (int r = 0; r < 8; ++r){
        int row = rowBase + waveM * 32 + i * 16 + r + halfAdd;
        int col = colBase + waveN * 64 + g * 16 + col0;
        float vv = acc[i][g][r];
        if (rowScale) vv *= rowScale[(size_t)row * scaleStride];
        size_t ci = (size_t)row * N + col;
        float outv = accumulate ? (C[ci] + vv) : vv;
        C[ci] = outv;
        if (Cbf) Cbf[ci] = f2bf(outv);
      }
    }
  }
}

// =====================================================================
// Flash attention: block = 64 q rows (4 waves x 16), 32-key tiles.
// q,k,v: bf16 [B*H, S, HD]; o: bf16 written into [NT, DM] at column h*HD.
// =====================================================================
__global__ __launch_bounds__(128)
void attn_kernel(const __bf16* __restrict__ q, const __bf16* __restrict__ k,
                 const __bf16* __restrict__ v, __bf16* __restrict__ o, int S)
{
  __shared__ __align__(16) __bf16 Qs[64 * 96];      // [row][d]
  __shared__ __align__(16) __bf16 Ks[32 * 96];      // [key][d]
  __shared__ __align__(16) __bf16 Vs[96 * 32];      // [d][key]
  __shared__ __align__(16) __bf16 Ps[4][16 * 32];   // per-wave P [row][key]

  const int tid  = threadIdx.x;
  const int wave = tid >> 5;
  const int lane = tid & 31;
  const int bh   = blockIdx.y;
  const int qBase = blockIdx.x * 64;
  const float scale = rsqrtf((float)HD);

  // async-stage Q tile (64 consecutive rows -> contiguous 6144 bf16)
  const __bf16* qp = q + ((size_t)bh * S + qBase) * HD;
#pragma unroll
  for (int j = 0; j < 6; ++j){
    int c = j * 128 + tid;            // 0..767 chunks of 8
    ASYNC_CP(qp + (size_t)c * 8, Qs + c * 8);
  }
  WAIT_ASYNC(0);
  __syncthreads();
  v16bf qa[3];
#pragma unroll
  for (int c = 0; c < 3; ++c) qa[c] = frag_a_ld(Qs + wave * 16 * 96 + c * 32, 96);

  float m_i[8], l_i[8];
  v8f oacc[6];
#pragma unroll
  for (int r = 0; r < 8; ++r){ m_i[r] = -1e30f; l_i[r] = 0.0f; }
#pragma unroll
  for (int j = 0; j < 6; ++j)
#pragma unroll
    for (int e = 0; e < 8; ++e) oacc[j][e] = 0.0f;

  for (int kt = 0; kt < S; kt += 32){
    __syncthreads();   // protect Ks/Vs WAR from previous tile
    const __bf16* kp = k + ((size_t)bh * S + kt) * HD;
    const __bf16* vp = v + ((size_t)bh * S + kt) * HD;
#pragma unroll
    for (int j = 0; j < 3; ++j){
      int c = j * 128 + tid;          // 0..383 chunks (32x96)
      ASYNC_CP(kp + (size_t)c * 8, Ks + c * 8);
    }
    // V staged transposed [d][key] (8-elem vector read, 8 scattered b16 stores)
#pragma unroll
    for (int j = 0; j < 3; ++j){
      int c = j * 128 + tid;
      int r = c / 12, dc = (c % 12) * 8;
      v8bf vv = *(const v8bf*)(vp + (size_t)r * HD + dc);
#pragma unroll
      for (int e = 0; e < 8; ++e) Vs[(dc + e) * 32 + r] = vv[e];
    }
    WAIT_ASYNC(0);
    __syncthreads();

    v8f sc0, sc1;
#pragma unroll
    for (int e = 0; e < 8; ++e){ sc0[e] = 0.0f; sc1[e] = 0.0f; }
#pragma unroll
    for (int c = 0; c < 3; ++c){
      sc0 = wmma_bf16(qa[c], frag_b_ld(Ks + c * 32, 96), sc0);
      sc1 = wmma_bf16(qa[c], frag_b_ld(Ks + 16 * 96 + c * 32, 96), sc1);
    }

    // online softmax (rows replicated across each 16-lane half)
#pragma unroll
    for (int r = 0; r < 8; ++r){
      float s0 = sc0[r] * scale;
      float s1 = sc1[r] * scale;
      float tmax = fmaxf(s0, s1);
#pragma unroll
      for (int off = 1; off < 16; off <<= 1)
        tmax = fmaxf(tmax, __shfl_xor(tmax, off, 16));
      float mn = fmaxf(m_i[r], tmax);
      float alpha = __expf(m_i[r] - mn);
      float p0 = __expf(s0 - mn);
      float p1 = __expf(s1 - mn);
      float ls = p0 + p1;
#pragma unroll
      for (int off = 1; off < 16; off <<= 1)
        ls += __shfl_xor(ls, off, 16);
      l_i[r] = l_i[r] * alpha + ls;
      m_i[r] = mn;
#pragma unroll
      for (int j = 0; j < 6; ++j) oacc[j][r] *= alpha;
      int prow = r + ((lane < 16) ? 0 : 8);
      Ps[wave][prow * 32 + (lane & 15)]      = f2bf(p0);
      Ps[wave][prow * 32 + 16 + (lane & 15)] = f2bf(p1);
    }
    asm volatile("s_wait_dscnt 0" ::: "memory");   // wave-local LDS RAW

    v16bf pa = frag_a_ld(&Ps[wave][0], 32);
#pragma unroll
    for (int j = 0; j < 6; ++j)
      oacc[j] = wmma_bf16(pa, frag_b_ld(Vs + j * 16 * 32, 32), oacc[j]);
  }

  const int b = bh / NH, h = bh - b * NH;
  const int halfAdd = (lane < 16) ? 0 : 8;
#pragma unroll
  for (int j = 0; j < 6; ++j){
#pragma unroll
    for (int r = 0; r < 8; ++r){
      int row = qBase + wave * 16 + r + halfAdd;
      int col = j * 16 + (lane & 15);
      o[((size_t)(b * S + row)) * DM + h * HD + col] = f2bf(oacc[j][r] / l_i[r]);
    }
  }
}

// ---------------- conversion / pointwise kernels ----------------

// W[K][N] f32 -> Wt[N][K] bf16  (tiled transpose, 32x32 tiles)
__global__ __launch_bounds__(256)
void transpose_bf16_kernel(const float* __restrict__ W, __bf16* __restrict__ Wt,
                           int K, int N)
{
  __shared__ __bf16 tile[32][33];
  const int bn = blockIdx.x * 32, bk = blockIdx.y * 32;
  const int tx = threadIdx.x & 31, ty = threadIdx.x >> 5;   // 32 x 8
#pragma unroll
  for (int j = 0; j < 32; j += 8)
    tile[ty + j][tx] = f2bf(W[(size_t)(bk + ty + j) * N + bn + tx]);
  __syncthreads();
#pragma unroll
  for (int j = 0; j < 32; j += 8)
    Wt[(size_t)(bn + ty + j) * K + bk + tx] = tile[tx][ty + j];
}

__global__ void rope_tables_kernel(float* __restrict__ cosT, float* __restrict__ sinT,
                                   const int* __restrict__ hPtr, const int* __restrict__ wPtr,
                                   const int* __restrict__ fPtr)
{
  int H = *hPtr, W = *wPtr; (void)fPtr;
  int p = blockIdx.x;
  int d = threadIdx.x;
  if (d >= HD) return;
  int t = p / (H * W), rem = p % (H * W);
  int yy = rem / W, xx = rem % W;
  int seg = d / 32, j = d % 32, jj = j & 15;
  float inv = __powf(10000.0f, -(2.0f * jj) / 32.0f);
  int nc = (seg == 0) ? xx : ((seg == 1) ? yy : t);
  float ang = (float)nc * inv;
  cosT[(size_t)p * HD + d] = __cosf(ang);
  sinT[(size_t)p * HD + d] = __sinf(ang);
}

// out = LN(x (+ y)) * g + b ; optional f32 and bf16 outputs
__global__ __launch_bounds__(256)
void add_layernorm_kernel(const float* __restrict__ x, const float* __restrict__ y,
                          const float* __restrict__ g, const float* __restrict__ b,
                          float* __restrict__ outF, __bf16* __restrict__ outB)
{
  __shared__ float red[256];
  const int n = blockIdx.x, t = threadIdx.x;
  float vals[3], s = 0.0f;
#pragma unroll
  for (int j = 0; j < 3; ++j){
    int d = t + j * 256;
    float v0 = x[(size_t)n * DM + d];
    if (y) v0 += y[(size_t)n * DM + d];
    vals[j] = v0; s += v0;
  }
  red[t] = s; __syncthreads();
  for (int off = 128; off > 0; off >>= 1){ if (t < off) red[t] += red[t + off]; __syncthreads(); }
  float mean = red[0] * (1.0f / DM);
  __syncthreads();
  float vs = 0.0f;
#pragma unroll
  for (int j = 0; j < 3; ++j){ float dd = vals[j] - mean; vs += dd * dd; }
  red[t] = vs; __syncthreads();
  for (int off = 128; off > 0; off >>= 1){ if (t < off) red[t] += red[t + off]; __syncthreads(); }
  float inv = rsqrtf(red[0] * (1.0f / DM) + 1e-5f);
#pragma unroll
  for (int j = 0; j < 3; ++j){
    int d = t + j * 256;
    float outv = (vals[j] - mean) * inv * g[d] + b[d];
    if (outF) outF[(size_t)n * DM + d] = outv;
    if (outB) outB[(size_t)n * DM + d] = f2bf(outv);
  }
}

// qkv [NT, 3*DM] f32 -> roped q,k and v, each bf16 [B*H, S, HD]
__global__ void rope_split_kernel(const float* __restrict__ qkv,
                                  const float* __restrict__ cosT, const float* __restrict__ sinT,
                                  __bf16* __restrict__ q, __bf16* __restrict__ k,
                                  __bf16* __restrict__ v)
{
  size_t idx = (size_t)blockIdx.x * 256 + threadIdx.x;
  if (idx >= (size_t)NT * DM) return;
  int d   = idx % HD;
  int tmp = idx / HD;
  int h   = tmp % NH;
  int n   = tmp / NH;
  int s   = n % SEQ, bb = n / SEQ;
  const float* row = qkv + (size_t)n * TD;
  int base = h * HD;
  float qv = row[base + d];
  float kv = row[DM + base + d];
  float vv = row[2 * DM + base + d];
  int rd = (d < 48) ? (d + 48) : (d - 48);
  float sgn = (d < 48) ? -1.0f : 1.0f;
  float qr = sgn * row[base + rd];
  float kr = sgn * row[DM + base + rd];
  float c = cosT[(size_t)s * HD + d], sn = sinT[(size_t)s * HD + d];
  size_t oidx = ((size_t)(bb * NH + h) * SEQ + s) * HD + d;
  q[oidx] = f2bf(qv * c + qr * sn);
  k[oidx] = f2bf(kv * c + kr * sn);
  v[oidx] = f2bf(vv);
}

// softmax gate + top-2 routing -> comb [NT, NE]
__global__ __launch_bounds__(64)
void router_kernel(const float* __restrict__ x, const float* __restrict__ gw,
                   float* __restrict__ comb)
{
  __shared__ float red[64 * NE];
  const int n = blockIdx.x, t = threadIdx.x;
  float a[NE] = {0.f, 0.f, 0.f, 0.f};
  for (int d = t; d < DM; d += 64){
    float xv = x[(size_t)n * DM + d];
#pragma unroll
    for (int e = 0; e < NE; ++e) a[e] += xv * gw[d * NE + e];
  }
#pragma unroll
  for (int e = 0; e < NE; ++e) red[t * NE + e] = a[e];
  __syncthreads();
  if (t == 0){
    float lg[NE];
#pragma unroll
    for (int e = 0; e < NE; ++e){
      float sum = 0.f;
      for (int i = 0; i < 64; ++i) sum += red[i * NE + e];
      lg[e] = sum;
    }
    float mx = lg[0];
#pragma unroll
    for (int e = 1; e < NE; ++e) mx = fmaxf(mx, lg[e]);
    float p[NE], ps = 0.f;
#pragma unroll
    for (int e = 0; e < NE; ++e){ p[e] = __expf(lg[e] - mx); ps += p[e]; }
#pragma unroll
    for (int e = 0; e < NE; ++e) p[e] /= ps;
    int i1 = 0;
#pragma unroll
    for (int e = 1; e < NE; ++e) if (p[e] > p[i1]) i1 = e;
    int i2 = -1;
#pragma unroll
    for (int e = 0; e < NE; ++e) if (e != i1 && (i2 < 0 || p[e] > p[i2])) i2 = e;
    float norm = p[i1] + p[i2] + 1e-5f;
#pragma unroll
    for (int e = 0; e < NE; ++e)
      comb[(size_t)n * NE + e] = ((e == i1) ? p[i1] : (e == i2) ? p[i2] : 0.0f) / norm;
  }
}

// act = bf16( silu(hg) * hu )
__global__ void silu_mul_kernel(const float* __restrict__ hg, const float* __restrict__ hu,
                                __bf16* __restrict__ act, size_t count)
{
  size_t i = (size_t)blockIdx.x * 256 + threadIdx.x;
  if (i >= count) return;
  float gv = hg[i];
  float sv = gv / (1.0f + __expf(-gv));
  act[i] = f2bf(sv * hu[i]);
}

// ---------------- host-side orchestration ----------------

static inline void launch_gemm(const __bf16* A, const __bf16* Bt, float* C, __bf16* Cbf,
                               const float* rowScale, int M, int N, int K,
                               int scaleStride, int accumulate, hipStream_t st)
{
  dim3 grid(N / 128, M / 128);
  gemm_kernel<<<grid, 256, 0, st>>>(A, Bt, C, Cbf, rowScale, M, N, K, scaleStride, accumulate);
}

static inline void launch_tr(const float* W, __bf16* Wt, int K, int N, hipStream_t st)
{
  dim3 grid(N / 32, K / 32);
  transpose_bf16_kernel<<<grid, 256, 0, st>>>(W, Wt, K, N);
}

extern "C" void kernel_launch(void* const* d_in, const int* in_sizes, int n_in,
                              void* d_out, int out_size, void* d_ws, size_t ws_size,
                              hipStream_t stream) {
  (void)in_sizes; (void)n_in; (void)out_size; (void)ws_size;

  const float* x_in  = (const float*)d_in[0];
  const int*   hPtr  = (const int*)d_in[1];
  const int*   wPtr  = (const int*)d_in[2];
  const int*   fPtr  = (const int*)d_in[3];
  const float* ln1_g = (const float*)d_in[4];
  const float* ln1_b = (const float*)d_in[5];
  const float* w_qkv = (const float*)d_in[6];
  const float* w_out = (const float*)d_in[7];
  const float* gatew = (const float*)d_in[8];
  const float* eg    = (const float*)d_in[9];
  const float* eu    = (const float*)d_in[10];
  const float* ed    = (const float*)d_in[11];
  const float* sg    = (const float*)d_in[12];
  const float* su    = (const float*)d_in[13];
  const float* sd    = (const float*)d_in[14];
  const float* ln2_g = (const float*)d_in[15];
  const float* ln2_b = (const float*)d_in[16];

  // ---- workspace arena ----
  float* ws = (float*)d_ws;
  size_t off = 0;
  auto allocF = [&](size_t nelem){ float* p = ws + off; off += nelem; return p; };
  float* cosT = allocF((size_t)SEQ * HD);
  float* sinT = allocF((size_t)SEQ * HD);
  float* qkvF = allocF((size_t)NT * TD);
  float* x2   = allocF((size_t)NT * DM);
  float* hg   = allocF((size_t)NT * FF);
  float* hu   = allocF((size_t)NT * FF);
  float* moe  = allocF((size_t)NT * DM);
  float* xcur = allocF((size_t)NT * DM);
  float* comb = allocF((size_t)NT * NE);

  __bf16* bb = (__bf16*)(ws + off);
  size_t boff = 0;
  auto allocB = [&](size_t nelem){ __bf16* p = bb + boff; boff += nelem; return p; };
  __bf16* xnB  = allocB((size_t)NT * DM);
  __bf16* oB   = allocB((size_t)NT * DM);
  __bf16* x2B  = allocB((size_t)NT * DM);
  __bf16* actB = allocB((size_t)NT * FF);
  __bf16* qB   = allocB((size_t)NT * DM);
  __bf16* kB   = allocB((size_t)NT * DM);
  __bf16* vB   = allocB((size_t)NT * DM);
  // per-layer bf16 transposed weights (reused across layers)
  __bf16* wqkvT = allocB((size_t)TD * DM);
  __bf16* woutT = allocB((size_t)DM * DM);
  __bf16* sgT   = allocB((size_t)FF * DM);
  __bf16* suT   = allocB((size_t)FF * DM);
  __bf16* sdT   = allocB((size_t)DM * FF);
  __bf16 *egT[NE], *euT[NE], *edT[NE];
  for (int e = 0; e < NE; ++e){
    egT[e] = allocB((size_t)FF * DM);
    euT[e] = allocB((size_t)FF * DM);
    edT[e] = allocB((size_t)DM * FF);
  }

  rope_tables_kernel<<<SEQ, HD, 0, stream>>>(cosT, sinT, hPtr, wPtr, fPtr);

  for (int i = 0; i < NL; ++i){
    const float* xin = (i == 0) ? x_in : xcur;

    // ---- convert this layer's weights to transposed bf16 ----
    launch_tr(w_qkv + (size_t)i * DM * TD, wqkvT, DM, TD, stream);
    launch_tr(w_out + (size_t)i * DM * DM, woutT, DM, DM, stream);
    launch_tr(sg + (size_t)i * DM * FF, sgT, DM, FF, stream);
    launch_tr(su + (size_t)i * DM * FF, suT, DM, FF, stream);
    launch_tr(sd + (size_t)i * FF * DM, sdT, FF, DM, stream);
    for (int e = 0; e < NE; ++e){
      size_t wo = (size_t)(i * NE + e);
      launch_tr(eg + wo * DM * FF, egT[e], DM, FF, stream);
      launch_tr(eu + wo * DM * FF, euT[e], DM, FF, stream);
      launch_tr(ed + wo * FF * DM, edT[e], FF, DM, stream);
    }

    // ---- attention ----
    add_layernorm_kernel<<<NT, 256, 0, stream>>>(xin, nullptr,
        ln1_g + (size_t)i * DM, ln1_b + (size_t)i * DM, nullptr, xnB);
    launch_gemm(xnB, wqkvT, qkvF, nullptr, nullptr, NT, TD, DM, 0, 0, stream);
    rope_split_kernel<<<(int)(((size_t)NT * DM + 255) / 256), 256, 0, stream>>>(
        qkvF, cosT, sinT, qB, kB, vB);
    attn_kernel<<<dim3(SEQ / 64, NB * NH), 128, 0, stream>>>(qB, kB, vB, oB, SEQ);
    (void)hipMemcpyAsync(x2, xin, (size_t)NT * DM * sizeof(float),
                         hipMemcpyDeviceToDevice, stream);
    launch_gemm(oB, woutT, x2, x2B, nullptr, NT, DM, DM, 0, 1, stream);

    // ---- MoE ----
    router_kernel<<<NT, 64, 0, stream>>>(x2, gatew + (size_t)i * DM * NE, comb);

    launch_gemm(x2B, sgT, hg, nullptr, nullptr, NT, FF, DM, 0, 0, stream);
    launch_gemm(x2B, suT, hu, nullptr, nullptr, NT, FF, DM, 0, 0, stream);
    silu_mul_kernel<<<(int)(((size_t)NT * FF + 255) / 256), 256, 0, stream>>>(
        hg, hu, actB, (size_t)NT * FF);
    launch_gemm(actB, sdT, moe, nullptr, nullptr, NT, DM, FF, 0, 0, stream);

    for (int e = 0; e < NE; ++e){
      launch_gemm(x2B, egT[e], hg, nullptr, nullptr, NT, FF, DM, 0, 0, stream);
      launch_gemm(x2B, euT[e], hu, nullptr, nullptr, NT, FF, DM, 0, 0, stream);
      silu_mul_kernel<<<(int)(((size_t)NT * FF + 255) / 256), 256, 0, stream>>>(
          hg, hu, actB, (size_t)NT * FF);
      launch_gemm(actB, edT[e], moe, nullptr, comb + e, NT, DM, FF, NE, 1, stream);
    }

    float* dst = (i == NL - 1) ? (float*)d_out : xcur;
    add_layernorm_kernel<<<NT, 256, 0, stream>>>(x2, moe,
        ln2_g + (size_t)i * DM, ln2_b + (size_t)i * DM, dst, nullptr);
  }
}
